// policy_network_19396072309467
// MI455X (gfx1250) — matrix operand order repaired and checked by reading, not compiled, not run on hardware
//
#include <hip/hip_runtime.h>

// ---------------------------------------------------------------------------
// NAS-cell RNN for MI455X (gfx1250, wave32, WMMA bf16 16x16x32).
// B=512, UNITS=256, T=256.  Each workgroup owns 16 batch rows and runs the
// entire 256-step recurrence with only intra-workgroup barriers.
// Weights are pre-packed to bf16 WMMA B-fragment layout in d_ws (L2-resident).
// ---------------------------------------------------------------------------

typedef __attribute__((ext_vector_type(16))) __bf16        v16bf;
typedef __attribute__((ext_vector_type(8)))  float         v8f;
typedef __attribute__((ext_vector_type(4)))  unsigned int  v4u;

#define UNITS   256
#define NCOL    2048      // 8 gates * 256
#define TSTEPS  256
#define HSTRIDE 264       // bf16 elements per h row in LDS (528 B, 16B-aligned, bank-spread)

union Frag {
    v4u   u[2];
    v16bf bf;
};

__device__ __forceinline__ unsigned short f2bf(float f) {
    unsigned int u = __builtin_bit_cast(unsigned int, f);
    u += 0x7FFFu + ((u >> 16) & 1u);          // round-to-nearest-even
    return (unsigned short)(u >> 16);
}

__device__ __forceinline__ float sigf(float x) {
    // 1 / (1 + e^-x)  via v_exp_f32 + v_rcp_f32
    return __builtin_amdgcn_rcpf(1.0f + __expf(-x));
}

__device__ __forceinline__ float tanh_fast(float x) {
    // 1 - 2/(e^{2x}+1): saturates cleanly to +/-1 without inf/inf
    float e = __expf(2.0f * x);
    return 1.0f - 2.0f * __builtin_amdgcn_rcpf(e + 1.0f);
}

// ---------------------------------------------------------------------------
// Prep: pack recurrent_kernel [256][2048] f32 -> bf16 WMMA B-fragments.
// Fragment (tile_n, kfrag) covers K = 32*kfrag..+31, N = 16*tile_n..+15.
// Layout per fragment (1024 B):
//   dwords  v=0..3 : byte  lane*16 + v*4
//   dwords  v=4..7 : byte  512 + lane*16 + (v-4)*4
// Lane L: N = L&15 ; K = 32*kfrag + (L<16 ? 0 : 16) + 2v, 2v+1 (lo,hi halves).
// grid: 1024 blocks (tile_n*8+kfrag) x 256 threads; 1 dword/thread.
// ---------------------------------------------------------------------------
__global__ __launch_bounds__(256) void pack_weights_bf16(
        const float* __restrict__ rk, unsigned int* __restrict__ packed) {
    const int frag   = blockIdx.x;        // 0..1023
    const int tile_n = frag >> 3;
    const int kfrag  = frag & 7;
    const int tid    = threadIdx.x;
    const int lane   = tid >> 3;          // 0..31
    const int v      = tid & 7;           // 0..7

    const int n     = (tile_n << 4) + (lane & 15);
    const int kbase = (kfrag << 5) + ((lane < 16) ? 0 : 16) + (v << 1);

    const float f0 = rk[(size_t)kbase * NCOL + n];
    const float f1 = rk[(size_t)(kbase + 1) * NCOL + n];
    const unsigned int dw = (unsigned int)f2bf(f0) | ((unsigned int)f2bf(f1) << 16);

    const int dword_in_frag = (v < 4) ? (lane * 4 + v) : (128 + lane * 4 + (v - 4));
    packed[frag * 256 + dword_in_frag] = dw;
}

// ---------------------------------------------------------------------------
// Main persistent recurrence kernel.  grid = 32 WGs x 256 threads (8 waves).
// Wave w owns column tiles (16g + 2w + jt) for g=0..7, jt=0..1.
// ---------------------------------------------------------------------------
__global__ __launch_bounds__(256) void nas_cell_rnn(
        const float* __restrict__ x,          // [512][256]
        const float* __restrict__ kern,       // [2048]
        const unsigned int* __restrict__ wpack,
        float* __restrict__ out) {            // [512][256]
    __shared__ __align__(16) unsigned short h_lds[16 * HSTRIDE]; // bf16 h, padded rows
    __shared__ float x_lds[16 * TSTEPS];                          // 16 KB x tile

    const int tid  = threadIdx.x;
    const int lane = tid & 31;
    const int w    = tid >> 5;        // wave id 0..7
    const int lh   = lane >> 4;       // lane-half (0/1)
    const int ln   = lane & 15;
    const int wg   = blockIdx.x;      // batch tile: rows wg*16 .. wg*16+15

    // init h = 0 in LDS (2112 dwords)
    for (int i = tid; i < (16 * HSTRIDE) / 2; i += 256)
        ((unsigned int*)h_lds)[i] = 0u;
    // stage x tile (contiguous rows wg*16..+15)
    for (int i = tid; i < 16 * TSTEPS; i += 256)
        x_lds[i] = x[(size_t)(wg * 16) * TSTEPS + i];

    // per-thread input-kernel weights: kw[g][jt] = kernel[256g + j]
    float kw[8][2];
#pragma unroll
    for (int g = 0; g < 8; ++g)
#pragma unroll
        for (int jt = 0; jt < 2; ++jt)
            kw[g][jt] = kern[g * 256 + w * 32 + jt * 16 + ln];

    // cell state c, resident in registers across all timesteps
    float c[2][8];
#pragma unroll
    for (int jt = 0; jt < 2; ++jt)
#pragma unroll
        for (int v = 0; v < 8; ++v) c[jt][v] = 0.0f;

    const v8f zero8 = {0.f, 0.f, 0.f, 0.f, 0.f, 0.f, 0.f, 0.f};
    v8f acc[16];                                  // acc[2g+jt]

    __syncthreads();

#pragma unroll 1
    for (int t = 0; t < TSTEPS; ++t) {
        // Opaque zero offset (in an SGPR): blocks LICM from hoisting the
        // (loop-invariant) B-fragment loads out of the t-loop -- which would
        // spill 1024 VGPRs of weights to scratch -- while keeping pointer
        // provenance through `wpack` intact so address-space inference still
        // emits global_load_b128 (saddr + imm offsets, LOADcnt only, no
        // flat/DScnt coupling) hitting the 192MB L2.
        unsigned int opq = 0;
        asm volatile("" : "+s"(opq));
        const v4u* wp4 = (const v4u*)(wpack + opq);

        // ------------------ GEMM: m = h @ rec_kernel (bf16 WMMA) -----------
#pragma unroll
        for (int i = 0; i < 16; ++i) acc[i] = zero8;

#pragma unroll
        for (int k = 0; k < 8; ++k) {
            // A fragment (shared across all 16 tiles): h rows, K = 32k..32k+31
            // lane: row = ln, K = 32k + 16*lh + s (contiguous, same order as packed B)
            const char* hb = (const char*)h_lds + ln * (HSTRIDE * 2) + k * 64 + lh * 32;
            Frag fa;
            fa.u[0] = *(const v4u*)(hb);          // K = 32k + (lh?16:0)  .. +7
            fa.u[1] = *(const v4u*)(hb + 16);     // K = 32k + 8 + (lh?16:0) .. +7

#pragma unroll
            for (int g = 0; g < 8; ++g) {
#pragma unroll
                for (int jt = 0; jt < 2; ++jt) {
                    const int tile = g * 16 + w * 2 + jt;     // global N-tile
                    const v4u* fb = wp4 + (((size_t)tile * 8 + k) << 6);
                    Frag b;
                    b.u[0] = fb[lane];            // dwords 0..3, coalesced
                    b.u[1] = fb[32 + lane];       // dwords 4..7, coalesced
                    acc[2 * g + jt] = __builtin_amdgcn_wmma_f32_16x16x32_bf16(
                        false, fa.bf, false, b.bf, (short)0, acc[2 * g + jt],
                        false, false);
                }
            }
        }

        __syncthreads();   // all h reads complete before anyone overwrites h

        // ------------------ NAS-cell elementwise wiring (pure VALU) --------
        float xv[8];
#pragma unroll
        for (int v = 0; v < 8; ++v)
            xv[v] = x_lds[(v + 8 * lh) * TSTEPS + t];

#pragma unroll
        for (int jt = 0; jt < 2; ++jt) {
            const int j = w * 32 + jt * 16 + ln;  // unit column 0..255
#pragma unroll
            for (int v = 0; v < 8; ++v) {
                const int M = v + 8 * lh;         // row within batch tile
                const float xb = xv[v];
                const float m0 = acc[0  + jt][v], m1 = acc[2  + jt][v];
                const float m2 = acc[4  + jt][v], m3 = acc[6  + jt][v];
                const float m4 = acc[8  + jt][v], m5 = acc[10 + jt][v];
                const float m6 = acc[12 + jt][v], m7 = acc[14 + jt][v];
                const float i0 = xb * kw[0][jt], i1 = xb * kw[1][jt];
                const float i2 = xb * kw[2][jt], i3 = xb * kw[3][jt];
                const float i4 = xb * kw[4][jt], i5 = xb * kw[5][jt];
                const float i6 = xb * kw[6][jt], i7 = xb * kw[7][jt];

                const float l1_0 = sigf(i0 + m0);
                const float l1_1 = fmaxf(i1 + m1, 0.0f);
                const float l1_2 = sigf(i2 + m2);
                const float l1_3 = fmaxf(i3 * m3, 0.0f);
                const float l1_4 = tanh_fast(i4 + m4);
                const float l1_5 = sigf(i5 + m5);
                const float l1_6 = tanh_fast(i6 + m6);
                const float l1_7 = sigf(i7 + m7);

                float l2_0 = tanh_fast(l1_0 * l1_1);
                const float l2_1 = tanh_fast(l1_2 + l1_3);
                const float l2_2 = tanh_fast(l1_4 * l1_5);
                const float l2_3 = sigf(l1_6 + l1_7);

                l2_0 = tanh_fast(l2_0 + c[jt][v]);
                const float nc   = l2_0 * l2_1;
                c[jt][v]         = nc;
                const float l3_1 = tanh_fast(l2_2 + l2_3);
                const float nh   = tanh_fast(nc * l3_1);

                h_lds[M * HSTRIDE + j] = f2bf(nh);
                if (t == TSTEPS - 1)
                    out[(size_t)(wg * 16 + M) * UNITS + j] = nh;
            }
        }

        __syncthreads();   // h fully written before next step's A-frag reads
    }
}

// ---------------------------------------------------------------------------
extern "C" void kernel_launch(void* const* d_in, const int* in_sizes, int n_in,
                              void* d_out, int out_size, void* d_ws, size_t ws_size,
                              hipStream_t stream) {
    (void)in_sizes; (void)n_in; (void)out_size; (void)ws_size;
    const float* x    = (const float*)d_in[0];   // [512][256]
    const float* kern = (const float*)d_in[1];   // [1][2048]
    const float* rk   = (const float*)d_in[2];   // [256][2048]
    unsigned int* wpack = (unsigned int*)d_ws;   // 1 MB packed bf16 fragments
    float* out = (float*)d_out;                  // [512][256]

    pack_weights_bf16<<<1024, 256, 0, stream>>>(rk, wpack);
    nas_cell_rnn<<<32, 256, 0, stream>>>(x, kern, (const unsigned int*)wpack, out);
}
